// DeformableAttention_24283745092395
// MI455X (gfx1250) — matrix operand order repaired and checked by reading, not compiled, not run on hardware
//
#include <hip/hip_runtime.h>

typedef _Float16 v16h __attribute__((ext_vector_type(16)));
typedef float    v8f  __attribute__((ext_vector_type(8)));
typedef _Float16 h8   __attribute__((ext_vector_type(8)));

#define C_DIM  256
#define HW_DIM 4096
#define B_DIM  4
#define W_IMG  64
#define NREF   4
#define BHWC   ((size_t)B_DIM * HW_DIM * C_DIM)

__device__ __forceinline__ float wave_sum(float v) {
#pragma unroll
  for (int m = 16; m > 0; m >>= 1) v += __shfl_xor(v, m, 32);
  return v;
}

__device__ __forceinline__ int iclamp(int v, int lo, int hi) {
  return v < lo ? lo : (v > hi ? hi : v);
}

// ---------------------------------------------------------------------------
// Stage 0a: pack Wq/Wk/Wv (f32 row-major 256x256) into f16 WMMA A-fragment
// order: [mat][mt(16)][kt(8)][lane(32)][h(16)], so stage 1 reads one
// contiguous v16h (32B) per lane per fragment.
// A layout used here (16x32, contiguous k, same order as B in stage 1): M = lane&15; g = lane>>4;
//   halves 0..15 -> K = g*16 + h
// ---------------------------------------------------------------------------
__global__ void pack_weights(const float* __restrict__ Wq,
                             const float* __restrict__ Wk,
                             const float* __restrict__ Wv,
                             _Float16* __restrict__ pW) {
  int idx = blockIdx.x * 256 + threadIdx.x;   // 3*65536 total
  int mat = idx >> 16;
  int r   = idx & 65535;
  int mt  = r >> 12;
  int r2  = r & 4095;
  int kt  = r2 >> 9;
  int r3  = r2 & 511;
  int lane = r3 >> 4;
  int h    = r3 & 15;
  int g = lane >> 4;
  int K = g * 16 + h;
  int row = mt * 16 + (lane & 15);
  int col = kt * 32 + K;
  const float* Wm = (mat == 0) ? Wq : (mat == 1) ? Wk : Wv;
  pW[idx] = (_Float16)Wm[row * C_DIM + col];
}

// ---------------------------------------------------------------------------
// Stage 0b: transpose x [B][C][HW] f32 -> xT [B][HW][C] f16 (32x32 LDS tiles)
// ---------------------------------------------------------------------------
__global__ void transpose_x(const float* __restrict__ x,
                            _Float16* __restrict__ xT) {
  __shared__ float tile[32][33];
  int bx  = blockIdx.x;
  int b   = bx >> 10;
  int rem = bx & 1023;
  int ct  = rem >> 7;    // 8 c-tiles
  int ht  = rem & 127;   // 128 hw-tiles
  int t = threadIdx.x, lane = t & 31, r = t >> 5;
  const float* xb = x + (size_t)b * C_DIM * HW_DIM;
#pragma unroll
  for (int j = 0; j < 4; ++j) {
    int cc = r * 4 + j;
    tile[cc][lane] = xb[(size_t)(ct * 32 + cc) * HW_DIM + ht * 32 + lane];
  }
  __syncthreads();
#pragma unroll
  for (int j = 0; j < 4; ++j) {
    int hw = ht * 32 + r * 4 + j;
    int c  = ct * 32 + lane;
    xT[((size_t)b * HW_DIM + hw) * C_DIM + c] = (_Float16)tile[lane][r * 4 + j];
  }
}

// ---------------------------------------------------------------------------
// Stage 1: Q/K/V projection. Block = 256 threads = 8 waves, 16 pixels.
// 48 output tiles (3 matrices x 16 row-tiles of 16 channels); wave w owns
// tiles {w, w+8, ..., w+40}. K = 256 -> 8 x v_wmma_f32_16x16x32_f16 per tile.
// B fragment (x tile) is shared across the 6 tiles per k-step.
// Results stored pixel-major f16: qkvT[mat][b][hw][c].
// ---------------------------------------------------------------------------
__global__ void proj_qkv(const _Float16* __restrict__ xT,
                         const _Float16* __restrict__ pW,
                         const float* __restrict__ bq,
                         const float* __restrict__ bk,
                         const float* __restrict__ bv,
                         _Float16* __restrict__ qkvT) {
  int blk = blockIdx.x;
  int b   = blk >> 8;     // 256 pixel-tiles per batch
  int pt  = blk & 255;
  int p0  = pt * 16;
  int t = threadIdx.x;
  int w = t >> 5, lane = t & 31;
  int ln = lane & 15, g = lane >> 4;

  // B fragment base: column N = ln (pixel p0+ln); halves = 16 consecutive c.
  const _Float16* xb = xT + ((size_t)b * HW_DIM + p0 + ln) * C_DIM + g * 16;

  v8f acc[6] = {};
  for (int kt = 0; kt < 8; ++kt) {
    v16h bf = *(const v16h*)(xb + kt * 32);
#pragma unroll
    for (int i = 0; i < 6; ++i) {
      int tid = w + 8 * i;
      int mat = tid >> 4, mt = tid & 15;
      v16h af = *(const v16h*)(pW +
          (size_t)(((mat * 16 + mt) * 8 + kt) * 32 + lane) * 16);
      acc[i] = __builtin_amdgcn_wmma_f32_16x16x32_f16(
          false, af, false, bf, (short)0, acc[i], false, false);
    }
  }

#pragma unroll
  for (int i = 0; i < 6; ++i) {
    int tid = w + 8 * i;
    int mat = tid >> 4, mt = tid & 15;
    const float* bias = (mat == 0) ? bq : (mat == 1) ? bk : bv;
    int c0 = mt * 16 + g * 8;   // lane holds 8 consecutive M rows
    h8 o;
#pragma unroll
    for (int rr = 0; rr < 8; ++rr)
      o[rr] = (_Float16)(acc[i][rr] + bias[c0 + rr]);
    *(h8*)(qkvT + (size_t)mat * BHWC +
           ((size_t)b * HW_DIM + p0 + ln) * C_DIM + c0) = o;
  }
}

// ---------------------------------------------------------------------------
// Stage 2: offsets (Woff @ Q), gather, softmax over n=4, attend, and
// transpose-store to [B][C][HW] through padded LDS.
// Block = 256 threads = 8 waves, 32 pixels; wave-per-pixel (4 each).
// ---------------------------------------------------------------------------
__global__ void attend(const _Float16* __restrict__ qkvT,
                       const float* __restrict__ Woff,
                       const float* __restrict__ boff,
                       float* __restrict__ out) {
  __shared__ float ot[32][257];   // stride 257 dwords -> conflict-free
  int b  = blockIdx.x >> 7;
  int pt = blockIdx.x & 127;
  int p0 = pt * 32;
  int t = threadIdx.x, w = t >> 5, lane = t & 31;
  const _Float16* Qt = qkvT;
  const _Float16* Kt = qkvT + BHWC;
  const _Float16* Vt = qkvT + 2 * BHWC;

  for (int k = 0; k < 4; ++k) {
    int pl = w * 4 + k;
    int hw = p0 + pl;

    h8 qv = *(const h8*)(Qt + ((size_t)b * HW_DIM + hw) * C_DIM + lane * 8);
    float q[8];
#pragma unroll
    for (int r = 0; r < 8; ++r) q[r] = (float)qv[r];

    // offsets[j] = Woff[j,:] . Q[:,pixel] + boff[j]
    float off[8];
#pragma unroll
    for (int j = 0; j < 8; ++j) {
      const float* wr = Woff + j * C_DIM + lane * 8;
      float s = 0.f;
#pragma unroll
      for (int r = 0; r < 8; ++r) s += wr[r] * q[r];
      off[j] = wave_sum(s) + boff[j];
    }

    int gw = hw & (W_IMG - 1), gh = hw >> 6;
    int idxn[NREF];
#pragma unroll
    for (int n = 0; n < NREF; ++n) {
      int rx = iclamp((int)rintf((float)gw + off[2 * n]),     0, W_IMG - 1);
      int ry = iclamp((int)rintf((float)gh + off[2 * n + 1]), 0, W_IMG - 1);
      idxn[n] = ry * W_IMG + rx;
    }

    float a[NREF];
#pragma unroll
    for (int n = 0; n < NREF; ++n) {
      h8 kv = *(const h8*)(Kt + ((size_t)b * HW_DIM + idxn[n]) * C_DIM + lane * 8);
      float s = 0.f;
#pragma unroll
      for (int r = 0; r < 8; ++r) s += q[r] * (float)kv[r];
      a[n] = wave_sum(s);
    }

    float m = fmaxf(fmaxf(a[0], a[1]), fmaxf(a[2], a[3]));
    float e[NREF], den = 0.f;
#pragma unroll
    for (int n = 0; n < NREF; ++n) { e[n] = __expf(a[n] - m); den += e[n]; }
    float inv = 1.0f / den;

    float o[8] = {};
#pragma unroll
    for (int n = 0; n < NREF; ++n) {
      h8 vv = *(const h8*)(Vt + ((size_t)b * HW_DIM + idxn[n]) * C_DIM + lane * 8);
      float wn = e[n] * inv;
#pragma unroll
      for (int r = 0; r < 8; ++r) o[r] += wn * (float)vv[r];
    }
#pragma unroll
    for (int r = 0; r < 8; ++r) ot[pl][lane * 8 + r] = o[r];
  }

  __syncthreads();

  // Transposed, coalesced store: lanes sweep consecutive hw for each channel.
#pragma unroll 4
  for (int i = 0; i < 32; ++i) {
    int c = w * 32 + i;
    out[((size_t)b * C_DIM + c) * HW_DIM + p0 + lane] = ot[lane][c];
  }
}

// ---------------------------------------------------------------------------
extern "C" void kernel_launch(void* const* d_in, const int* in_sizes, int n_in,
                              void* d_out, int out_size, void* d_ws, size_t ws_size,
                              hipStream_t stream) {
  const float* x    = (const float*)d_in[0];
  const float* Wq   = (const float*)d_in[1];
  const float* bq   = (const float*)d_in[2];
  const float* Wk   = (const float*)d_in[3];
  const float* bk   = (const float*)d_in[4];
  const float* Wv   = (const float*)d_in[5];
  const float* bv   = (const float*)d_in[6];
  const float* Woff = (const float*)d_in[7];
  const float* boff = (const float*)d_in[8];
  float* out = (float*)d_out;

  char* ws = (char*)d_ws;
  _Float16* pW   = (_Float16*)ws;                        // 3*65536*2  = 384 KB
  _Float16* xT   = (_Float16*)(ws + 393216);             // B*HW*C*2   = 8 MB
  _Float16* qkvT = (_Float16*)(ws + 393216 + 8388608);   // 3*B*HW*C*2 = 24 MB

  pack_weights<<<768, 256, 0, stream>>>(Wq, Wk, Wv, pW);
  transpose_x<<<4096, 256, 0, stream>>>(x, xT);
  proj_qkv<<<1024, 256, 0, stream>>>(xT, pW, bq, bk, bv, qkvT);
  attend<<<512, 256, 0, stream>>>(qkvT, Woff, boff, out);
}
